// GraphConv_21165598835037
// MI455X (gfx1250) — matrix operand order repaired and checked by reading, not compiled, not run on hardware
//
#include <hip/hip_runtime.h>
#include <hip/hip_bf16.h>
#include <math.h>

typedef __attribute__((ext_vector_type(16))) _Float16 v16h;
typedef __attribute__((ext_vector_type(8)))  _Float16 v8h;
typedef __attribute__((ext_vector_type(8)))  float    v8f;

#define LDS_STRIDE 136   // halfs per row; multiple of 8 -> keeps 16B alignment, spreads banks

// ---------------------------------------------------------------------------
// WMMA helpers (CDNA5 wave32, v_wmma_f32_16x16x32_f16)
// ---------------------------------------------------------------------------
__device__ __forceinline__ v8f wmma_f16(v16h a, v16h b, v8f c) {
  return __builtin_amdgcn_wmma_f32_16x16x32_f16(false, a, false, b, (short)0, c,
                                                false, false);
}

// A-matrix (16x32 f16) lane layout per ISA:
// lanes 0-15: m=lane, K pairs 0..7 (VGPR0-3) and 16..23 (VGPR4-7)
// lanes16-31: m=lane-16, K pairs 8..15 and 24..31
// => per lane: two contiguous 8-half chunks at k = s*32+base0 and s*32+16+base0
__device__ __forceinline__ v16h load_a16(const _Float16* Xrow, int s, int base0) {
  const v8h lo = *(const v8h*)(Xrow + s * 32 + base0);
  const v8h hi = *(const v8h*)(Xrow + s * 32 + 16 + base0);
  v16h a;
#pragma unroll
  for (int i = 0; i < 8; ++i) { a[i] = lo[i]; a[i + 8] = hi[i]; }
  return a;
}

// ---------------------------------------------------------------------------
// Kernel 0: repack fp32 [128,128] weights (y = x @ W) into f16 WMMA B panels.
// Panel layout: [coltile c(8)][kstep s(4)][lane(32)][16 f16]  (32B per lane).
// B (32x16): lane n = lane&15; lanes 0-15 cover K 0..7,16..23, lanes 16-31 K 8..15,24..31;
// pair p holds K = s*32 + (lane>>4)*8 + (p>>2)*16 + 2(p&3), +1 of column n.
// 13 matrices: K/Q/V x 3 types (0..8), A x 3 types (9..11), s2u (12).
// ---------------------------------------------------------------------------
__global__ __launch_bounds__(256) void pack_weights_kernel(
    const float* __restrict__ Kw, const float* __restrict__ Qw,
    const float* __restrict__ Vw, const float* __restrict__ Aw,
    const float* __restrict__ s2u, _Float16* __restrict__ packed) {
  int b = blockIdx.x;  // 0..12
  const float* W;
  if (b < 9) {
    const float* base = (b / 3 == 0) ? Kw : (b / 3 == 1) ? Qw : Vw;
    W = base + (size_t)(b % 3) * 128 * 128;
  } else if (b < 12) {
    W = Aw + (size_t)(b - 9) * 128 * 128;
  } else {
    W = s2u;
  }
  _Float16* dst = packed + (size_t)b * 16384;
  for (int i = threadIdx.x; i < 8 * 4 * 32; i += blockDim.x) {
    int c = i >> 7;           // /128
    int s = (i >> 5) & 3;
    int lane = i & 31;
    int n = c * 16 + (lane & 15);
    int khalf = lane >> 4;
    _Float16* o = dst + (size_t)i * 16;
#pragma unroll
    for (int p = 0; p < 8; ++p) {
      int k = s * 32 + khalf * 8 + (p >> 2) * 16 + 2 * (p & 3);
      o[2 * p]     = (_Float16)W[(size_t)k * 128 + n];
      o[2 * p + 1] = (_Float16)W[(size_t)(k + 1) * 128 + n];
    }
  }
}

// ---------------------------------------------------------------------------
// Kernel 1: per-type K/Q/V projections via WMMA; per-row type selection.
// Block = 256 thr (8 waves) handles a 16-node tile; wave w owns out-coltile w.
// Also emits v1 = V-projection with type-1 weights for ALL nodes (speaker path).
// ---------------------------------------------------------------------------
__global__ __launch_bounds__(256) void proj_kernel(
    const float* __restrict__ xin, const int* __restrict__ ntype,
    const _Float16* __restrict__ packed, const float* __restrict__ Kb,
    const float* __restrict__ Qb, const float* __restrict__ Vb,
    float* __restrict__ kout, float* __restrict__ qout,
    float* __restrict__ vout, float* __restrict__ v1out, int N) {
  __shared__ _Float16 X[16 * LDS_STRIDE];
  __shared__ int ty[16];
  const int tile0 = blockIdx.x * 16;
  const int tid = threadIdx.x;

  for (int i = tid; i < 16 * 128; i += 256) {
    int r = i >> 7, c = i & 127;
    int row = tile0 + r;
    float v = (row < N) ? xin[(size_t)row * 128 + c] : 0.f;
    X[r * LDS_STRIDE + c] = (_Float16)v;
  }
  if (tid < 16) ty[tid] = (tile0 + tid < N) ? ntype[tile0 + tid] : 0;
  __syncthreads();

  const int lane = tid & 31;
  const int cT = tid >> 5;                 // coltile 0..7
  const int m = lane & 15;
  const int base0 = (lane >> 4) * 8;
  const _Float16* Xrow = &X[m * LDS_STRIDE];
  const int col = cT * 16 + (lane & 15);
  const int rbase = (lane >> 4) * 8;

  for (int mat = 0; mat < 3; ++mat) {
    v8f acc0 = {}, acc1 = {}, acc2 = {};
    const _Float16* p0 = packed + (size_t)(mat * 3 + 0) * 16384;
    const _Float16* p1 = packed + (size_t)(mat * 3 + 1) * 16384;
    const _Float16* p2 = packed + (size_t)(mat * 3 + 2) * 16384;
#pragma unroll
    for (int s = 0; s < 4; ++s) {
      v16h a = load_a16(Xrow, s, base0);
      size_t bo = (size_t)((cT * 4 + s) * 32 + lane) * 16;
      v16h b0 = *(const v16h*)(p0 + bo);
      v16h b1 = *(const v16h*)(p1 + bo);
      v16h b2 = *(const v16h*)(p2 + bo);
      acc0 = wmma_f16(a, b0, acc0);
      acc1 = wmma_f16(a, b1, acc1);
      acc2 = wmma_f16(a, b2, acc2);
    }
    const float* Bv = (mat == 0) ? Kb : (mat == 1) ? Qb : Vb;
    float* Out = (mat == 0) ? kout : (mat == 1) ? qout : vout;
    float bia0 = Bv[col], bia1 = Bv[128 + col], bia2 = Bv[256 + col];
    float a0[8], a1[8], a2[8];
#pragma unroll
    for (int j = 0; j < 8; ++j) { a0[j] = acc0[j]; a1[j] = acc1[j]; a2[j] = acc2[j]; }
#pragma unroll
    for (int j = 0; j < 8; ++j) {
      int r = rbase + j;
      int row = tile0 + r;
      if (row >= N) continue;
      int t = ty[r];
      float val = (t == 0) ? (a0[j] + bia0) : (t == 1) ? (a1[j] + bia1) : (a2[j] + bia2);
      Out[(size_t)row * 128 + col] = val;
      if (mat == 2) v1out[(size_t)row * 128 + col] = a1[j] + bia1;
    }
  }
}

// ---------------------------------------------------------------------------
// Kernel 2: su = v1 @ relation_s2u (single packed matrix, no bias) via WMMA.
// ---------------------------------------------------------------------------
__global__ __launch_bounds__(256) void s2u_kernel(
    const float* __restrict__ v1, const _Float16* __restrict__ pmat,
    float* __restrict__ su, int N) {
  __shared__ _Float16 X[16 * LDS_STRIDE];
  const int tile0 = blockIdx.x * 16;
  const int tid = threadIdx.x;
  for (int i = tid; i < 16 * 128; i += 256) {
    int r = i >> 7, c = i & 127;
    int row = tile0 + r;
    float v = (row < N) ? v1[(size_t)row * 128 + c] : 0.f;
    X[r * LDS_STRIDE + c] = (_Float16)v;
  }
  __syncthreads();
  const int lane = tid & 31;
  const int cT = tid >> 5;
  const _Float16* Xrow = &X[(lane & 15) * LDS_STRIDE];
  const int base0 = (lane >> 4) * 8;
  v8f acc = {};
#pragma unroll
  for (int s = 0; s < 4; ++s) {
    v16h a = load_a16(Xrow, s, base0);
    v16h b = *(const v16h*)(pmat + (size_t)((cT * 4 + s) * 32 + lane) * 16);
    acc = wmma_f16(a, b, acc);
  }
  const int col = cT * 16 + (lane & 15);
  const int rbase = (lane >> 4) * 8;
  float ar[8];
#pragma unroll
  for (int j = 0; j < 8; ++j) ar[j] = acc[j];
#pragma unroll
  for (int j = 0; j < 8; ++j) {
    int row = tile0 + rbase + j;
    if (row < N) su[(size_t)row * 128 + col] = ar[j];
  }
}

// ---------------------------------------------------------------------------
// Kernel 3: per-edge attention logits + float atomic-max via ordered uint key.
// Wave per edge: lane = (head h = lane>>2, 4 output dims f0 = (lane&3)*4).
// ---------------------------------------------------------------------------
__device__ __forceinline__ unsigned float_order_key(float x) {
  unsigned u = __float_as_uint(x);
  return (u & 0x80000000u) ? ~u : (u | 0x80000000u);
}

__global__ __launch_bounds__(256) void edge_att_kernel(
    const int* __restrict__ ei, const int* __restrict__ etype,
    const int* __restrict__ ntype, const float* __restrict__ kbuf,
    const float* __restrict__ qbuf, const float* __restrict__ rel_att,
    const float* __restrict__ rel_pri, float* __restrict__ attraw,
    unsigned* __restrict__ mkey, int E) {
  int e = blockIdx.x * 8 + (threadIdx.x >> 5);
  if (e >= E) return;
  const int lane = threadIdx.x & 31;
  const int src = ei[e], tgt = ei[E + e], r = etype[e];
  const int h = lane >> 2, f0 = (lane & 3) * 4;

  // Each lane loads 4 elems of k_j[h,:]; broadcast within head quad via shfl.
  const float* kj = kbuf + (size_t)src * 128 + h * 16;
  float kreg[4];
#pragma unroll
  for (int i = 0; i < 4; ++i) kreg[i] = kj[f0 + i];

  const float* Wm = rel_att + (size_t)(r * 8 + h) * 256;  // [16][16]
  float kt0 = 0.f, kt1 = 0.f, kt2 = 0.f, kt3 = 0.f;
#pragma unroll
  for (int d = 0; d < 16; ++d) {
    float kv = __shfl(kreg[d & 3], (h << 2) + (d >> 2), 32);
    const float* wr = Wm + d * 16 + f0;
    kt0 += kv * wr[0]; kt1 += kv * wr[1]; kt2 += kv * wr[2]; kt3 += kv * wr[3];
  }
  const float* qi = qbuf + (size_t)tgt * 128 + h * 16 + f0;
  float part = qi[0] * kt0 + qi[1] * kt1 + qi[2] * kt2 + qi[3] * kt3;
  part += __shfl_xor(part, 1, 32);
  part += __shfl_xor(part, 2, 32);
  if ((lane & 3) == 0) {
    int st = ntype[src], tt = ntype[tgt];
    float pri = rel_pri[((size_t)(tt * 4 + r) * 3 + st) * 8 + h];
    float araw = part * pri * 0.25f;  // 1/sqrt(DK), DK=16
    attraw[(size_t)e * 8 + h] = araw;
    atomicMax(mkey + (size_t)tgt * 8 + h, float_order_key(araw));
  }
}

// ---------------------------------------------------------------------------
// Kernel 4: ex = exp(att - max[tgt]); den[tgt] += ex (ex overwrites attraw).
// ---------------------------------------------------------------------------
__global__ __launch_bounds__(256) void edge_softmax_kernel(
    const int* __restrict__ ei, float* __restrict__ attraw,
    const unsigned* __restrict__ mkey, float* __restrict__ den, int E) {
  int i = blockIdx.x * blockDim.x + threadIdx.x;
  if (i >= E * 8) return;
  int e = i >> 3, h = i & 7;
  int tgt = ei[E + e];
  unsigned u = mkey[(size_t)tgt * 8 + h];
  float mval = 0.f;
  if (u != 0u) {
    unsigned b = (u & 0x80000000u) ? (u ^ 0x80000000u) : ~u;
    mval = __uint_as_float(b);
    if (!isfinite(mval)) mval = 0.f;
  }
  float ex = __expf(attraw[i] - mval);
  attraw[i] = ex;
  atomicAdd(den + (size_t)tgt * 8 + h, ex);
}

// ---------------------------------------------------------------------------
// Kernel 5: message transform + weighted scatter-add; speaker-info writes.
// ---------------------------------------------------------------------------
__global__ __launch_bounds__(256) void edge_aggr_kernel(
    const int* __restrict__ ei, const int* __restrict__ etype,
    const float* __restrict__ vbuf, const float* __restrict__ rel_msg,
    const float* __restrict__ ex, const float* __restrict__ den,
    const float* __restrict__ su, float* __restrict__ aggr,
    float* __restrict__ spk, int E) {
  int e = blockIdx.x * 8 + (threadIdx.x >> 5);
  if (e >= E) return;
  const int lane = threadIdx.x & 31;
  const int src = ei[e], tgt = ei[E + e], r = etype[e];
  const int h = lane >> 2, f0 = (lane & 3) * 4;

  const float* vj = vbuf + (size_t)src * 128 + h * 16;
  float vreg[4];
#pragma unroll
  for (int i = 0; i < 4; ++i) vreg[i] = vj[f0 + i];

  const float* Wm = rel_msg + (size_t)(r * 8 + h) * 256;
  float vt0 = 0.f, vt1 = 0.f, vt2 = 0.f, vt3 = 0.f;
#pragma unroll
  for (int d = 0; d < 16; ++d) {
    float vv = __shfl(vreg[d & 3], (h << 2) + (d >> 2), 32);
    const float* wr = Wm + d * 16 + f0;
    vt0 += vv * wr[0]; vt1 += vv * wr[1]; vt2 += vv * wr[2]; vt3 += vv * wr[3];
  }
  float att = ex[(size_t)e * 8 + h] / (den[(size_t)tgt * 8 + h] + 1e-16f);
  float* dst = aggr + (size_t)tgt * 128 + h * 16 + f0;
  atomicAdd(dst + 0, vt0 * att);
  atomicAdd(dst + 1, vt1 * att);
  atomicAdd(dst + 2, vt2 * att);
  atomicAdd(dst + 3, vt3 * att);

  if (r == 0) {  // speaker info: spk[src] = su[tgt] (.set semantics)
    const float* s = su + (size_t)tgt * 128 + lane * 4;
    float* d = spk + (size_t)src * 128 + lane * 4;
    d[0] = s[0]; d[1] = s[1]; d[2] = s[2]; d[3] = s[3];
  }
}

// ---------------------------------------------------------------------------
// Kernel 6: gelu(aggr+spk) -> per-type A-projection (WMMA) -> skip blend.
// ---------------------------------------------------------------------------
__global__ __launch_bounds__(256) void out_kernel(
    const float* __restrict__ aggr, const float* __restrict__ spk,
    const float* __restrict__ xin, const int* __restrict__ ntype,
    const _Float16* __restrict__ packedA, const float* __restrict__ Ab,
    const float* __restrict__ skip, float* __restrict__ out, int N) {
  __shared__ _Float16 X[16 * LDS_STRIDE];
  __shared__ int ty[16];
  const int tile0 = blockIdx.x * 16;
  const int tid = threadIdx.x;
  for (int i = tid; i < 16 * 128; i += 256) {
    int r = i >> 7, c = i & 127;
    int row = tile0 + r;
    float g = 0.f;
    if (row < N) {
      float x = aggr[(size_t)row * 128 + c] + spk[(size_t)row * 128 + c];
      g = 0.5f * x * (1.0f + erff(x * 0.7071067811865475f));  // exact GELU
    }
    X[r * LDS_STRIDE + c] = (_Float16)g;
  }
  if (tid < 16) ty[tid] = (tile0 + tid < N) ? ntype[tile0 + tid] : 0;
  __syncthreads();

  const int lane = tid & 31;
  const int cT = tid >> 5;
  const _Float16* Xrow = &X[(lane & 15) * LDS_STRIDE];
  const int base0 = (lane >> 4) * 8;
  v8f acc0 = {}, acc1 = {}, acc2 = {};
  const _Float16* p0 = packedA + (size_t)0 * 16384;
  const _Float16* p1 = packedA + (size_t)1 * 16384;
  const _Float16* p2 = packedA + (size_t)2 * 16384;
#pragma unroll
  for (int s = 0; s < 4; ++s) {
    v16h a = load_a16(Xrow, s, base0);
    size_t bo = (size_t)((cT * 4 + s) * 32 + lane) * 16;
    acc0 = wmma_f16(a, *(const v16h*)(p0 + bo), acc0);
    acc1 = wmma_f16(a, *(const v16h*)(p1 + bo), acc1);
    acc2 = wmma_f16(a, *(const v16h*)(p2 + bo), acc2);
  }
  const int col = cT * 16 + (lane & 15);
  const int rbase = (lane >> 4) * 8;
  float bia0 = Ab[col], bia1 = Ab[128 + col], bia2 = Ab[256 + col];
  float al0 = 1.f / (1.f + __expf(-skip[0]));
  float al1 = 1.f / (1.f + __expf(-skip[1]));
  float al2 = 1.f / (1.f + __expf(-skip[2]));
  float a0[8], a1[8], a2[8];
#pragma unroll
  for (int j = 0; j < 8; ++j) { a0[j] = acc0[j]; a1[j] = acc1[j]; a2[j] = acc2[j]; }
#pragma unroll
  for (int j = 0; j < 8; ++j) {
    int row = tile0 + rbase + j;
    if (row >= N) continue;
    int t = ty[rbase + j];
    float val = (t == 0) ? (a0[j] + bia0) : (t == 1) ? (a1[j] + bia1) : (a2[j] + bia2);
    float alpha = (t == 0) ? al0 : (t == 1) ? al1 : al2;
    out[(size_t)row * 128 + col] =
        val * alpha + xin[(size_t)row * 128 + col] * (1.0f - alpha);
  }
}

// ---------------------------------------------------------------------------
extern "C" void kernel_launch(void* const* d_in, const int* in_sizes, int n_in,
                              void* d_out, int out_size, void* d_ws, size_t ws_size,
                              hipStream_t stream) {
  const float* node_inp  = (const float*)d_in[0];
  const int*   node_type = (const int*)d_in[1];
  const int*   edge_idx  = (const int*)d_in[2];
  const int*   edge_type = (const int*)d_in[3];
  // d_in[4] node_position: unused by the reference computation
  const float* Kw = (const float*)d_in[5];
  const float* Kb = (const float*)d_in[6];
  const float* Qw = (const float*)d_in[7];
  const float* Qb = (const float*)d_in[8];
  const float* Vw = (const float*)d_in[9];
  const float* Vb = (const float*)d_in[10];
  const float* Aw = (const float*)d_in[11];
  const float* Ab = (const float*)d_in[12];
  const float* rel_pri = (const float*)d_in[13];
  const float* rel_att = (const float*)d_in[14];
  const float* rel_msg = (const float*)d_in[15];
  const float* s2u     = (const float*)d_in[16];
  const float* skip    = (const float*)d_in[17];

  const int N = in_sizes[0] / 128;
  const int E = in_sizes[3];

  // Workspace carve-up (256B aligned)
  char* base = (char*)d_ws;
  size_t off = 0;
  auto carve = [&](size_t bytes) -> char* {
    char* p = base + off;
    off = (off + bytes + 255) & ~(size_t)255;
    return p;
  };
  _Float16* packed = (_Float16*)carve((size_t)13 * 16384 * sizeof(_Float16));
  float* kbuf  = (float*)carve((size_t)N * 128 * 4);
  float* qbuf  = (float*)carve((size_t)N * 128 * 4);
  float* vbuf  = (float*)carve((size_t)N * 128 * 4);
  float* v1buf = (float*)carve((size_t)N * 128 * 4);
  float* subuf = (float*)carve((size_t)N * 128 * 4);
  float* attraw = (float*)carve((size_t)E * 8 * 4);   // reused as ex
  unsigned* mkey = (unsigned*)carve((size_t)N * 8 * 4);
  float* den  = (float*)carve((size_t)N * 8 * 4);
  float* aggr = (float*)carve((size_t)N * 128 * 4);
  float* spk  = (float*)carve((size_t)N * 128 * 4);

  // Zero-init accumulators (graph-capture safe)
  hipMemsetAsync(mkey, 0, (size_t)N * 8 * 4, stream);
  hipMemsetAsync(den,  0, (size_t)N * 8 * 4, stream);
  hipMemsetAsync(aggr, 0, (size_t)N * 128 * 4, stream);
  hipMemsetAsync(spk,  0, (size_t)N * 128 * 4, stream);

  const int ntiles = (N + 15) / 16;

  pack_weights_kernel<<<13, 256, 0, stream>>>(Kw, Qw, Vw, Aw, s2u, packed);

  proj_kernel<<<ntiles, 256, 0, stream>>>(node_inp, node_type, packed,
                                          Kb, Qb, Vb, kbuf, qbuf, vbuf, v1buf, N);

  s2u_kernel<<<ntiles, 256, 0, stream>>>(v1buf, packed + (size_t)12 * 16384,
                                         subuf, N);

  edge_att_kernel<<<(E + 7) / 8, 256, 0, stream>>>(edge_idx, edge_type, node_type,
                                                   kbuf, qbuf, rel_att, rel_pri,
                                                   attraw, mkey, E);

  edge_softmax_kernel<<<(E * 8 + 255) / 256, 256, 0, stream>>>(edge_idx, attraw,
                                                               mkey, den, E);

  edge_aggr_kernel<<<(E + 7) / 8, 256, 0, stream>>>(edge_idx, edge_type, vbuf,
                                                    rel_msg, attraw, den, subuf,
                                                    aggr, spk, E);

  out_kernel<<<ntiles, 256, 0, stream>>>(aggr, spk, node_inp, node_type,
                                         packed + (size_t)9 * 16384, Ab, skip,
                                         (float*)d_out, N);
}
